// FTE_4810363372654
// MI455X (gfx1250) — matrix operand order repaired and checked by reading, not compiled, not run on hardware
//
#include <hip/hip_runtime.h>
#include <hip/hip_bf16.h>
#include <math.h>

// ---------------------------------------------------------------------------
// Fused equivariant-update kernel for MI455X (gfx1250, wave32, WMMA).
// All GEMMs run on v_wmma_f32_16x16x32_bf16 with f32 accumulation.
// ---------------------------------------------------------------------------

typedef __attribute__((ext_vector_type(16))) __bf16 v16bf;
typedef __attribute__((ext_vector_type(8)))  __bf16 v8bf;
typedef __attribute__((ext_vector_type(8)))  float  v8f;

#define H      256
#define FH1    48
#define FH2    8
#define NPB    16          // nodes per block
#define THREADS 256
#define NWAVES  8

// ---- LDS layout (bytes), regions reused across pipeline phases -------------
#define ST48  260          // f32 stride (256 + 4 pad)
#define STA   264          // bf16 stride (256 + 8 pad)
#define STXH  520          // bf16 stride (512 + 8 pad)

#define OFF_VEC2 0                          // vec2 f32 [48][260]     (phase2..6)
#define SZ_VEC2  (48 * ST48 * 4)            // 49920
#define OFF_VEC1 (OFF_VEC2 + SZ_VEC2)       // vec1 f32 [48][260]     (phase2..3)
#define SZ_VEC1  (48 * ST48 * 4)            // 49920
#define OFF_T1S  OFF_VEC1                   // per-wave t1 scratch    (phase4) aliases vec1
#define OFF_X    (OFF_VEC1 + SZ_VEC1)       // A-tile / s / h1 region (aliased)
#define SZ_X     32768                      // max(25344, 32768, 8448)
#define OFF_VDOT (OFF_X + SZ_X)             // vec_dot f32 [16][260]  (phase3..6)
#define SZ_VDOT  (16 * ST48 * 4)            // 16640
#define OFF_XH   (OFF_VDOT + SZ_VDOT)       // xh bf16 [16][520]      (phase0..5)
#define SZ_XH    (16 * STXH * 2)            // 16640
#define LDS_BYTES (OFF_XH + SZ_XH)          // 165888 (< 320KB WGP LDS)

static_assert(LDS_BYTES <= 320 * 1024, "LDS budget");

// ---------------------------------------------------------------------------

__device__ __forceinline__ v8f zero8() {
    v8f z;
#pragma unroll
    for (int i = 0; i < 8; ++i) z[i] = 0.f;
    return z;
}

__device__ __forceinline__ float silu(float x) { return x / (1.f + __expf(-x)); }

__device__ __forceinline__ v16bf cat16(v8bf lo, v8bf hi) {
    return __builtin_shufflevector(lo, hi, 0, 1, 2, 3, 4, 5, 6, 7,
                                          8, 9, 10, 11, 12, 13, 14, 15);
}

// A fragment (16x32 bf16) from row-major LDS tile.
// Layout (ISA 7.12.2): lanes 0-15 row=lane, K 0-7 & 16-23; lanes 16-31 row=lane-16,
// K 8-15 & 24-31.  -> per lane two contiguous 16B chunks.
__device__ __forceinline__ v16bf load_a_lds(const __bf16* base, int stride, int lane) {
    int h = lane >> 4, r = lane & 15;
    const __bf16* p = base + r * stride + h * 8;
    v8bf lo = *(const v8bf*)p;
    v8bf hi = *(const v8bf*)(p + 16);
    return cat16(lo, hi);
}

// B fragment (32x16 bf16) from global weights stored N-major (row = out col, K elems).
// Layout: lane 0-15 col=lane K 0-7 & 16-23; lane 16-31 col=lane-16 K 8-15 & 24-31.
__device__ __forceinline__ v16bf load_b_glb(const __bf16* __restrict__ wt, int K,
                                            int n0, int k0, int lane) {
    int n = lane & 15, kh = lane >> 4;
    const __bf16* p = wt + (size_t)(n0 + n) * K + (k0 + kh * 8);
    v8bf lo = *(const v8bf*)p;
    v8bf hi = *(const v8bf*)(p + 16);
    return cat16(lo, hi);
}

#define WMMA_BF16(a, b, c) \
    __builtin_amdgcn_wmma_f32_16x16x32_bf16(false, (a), false, (b), (short)0, (c), false, false)

// ---------------------------------------------------------------------------
// Prep: transpose f32 weight (Kin x Nin, row-major) -> bf16 N-major (Npad x Kpad),
// zero padding outside the valid region. Weights end up L2-resident (<1MB total).
// ---------------------------------------------------------------------------
__global__ void k_transpose_pad(const float* __restrict__ src, __bf16* __restrict__ dst,
                                int Kin, int Nin, int Kpad, int Npad) {
    int idx = blockIdx.x * blockDim.x + threadIdx.x;
    int total = Kpad * Npad;
    if (idx >= total) return;
    int k = idx % Kpad;
    int o = idx / Kpad;
    float v = (k < Kin && o < Nin) ? src[(size_t)k * Nin + o] : 0.f;
    dst[idx] = (__bf16)v;
}

// ---------------------------------------------------------------------------
// Fused main kernel: one block = 16 nodes through the full pipeline.
// ---------------------------------------------------------------------------
__global__ __launch_bounds__(THREADS, 1)
void fte_fused(const float* __restrict__ x, const float* __restrict__ vec,
               const float* __restrict__ frame,
               const __bf16* __restrict__ WtE, const __bf16* __restrict__ Wt1,
               const __bf16* __restrict__ Wt2, const __bf16* __restrict__ WtA,
               const __bf16* __restrict__ WtB,
               const float* __restrict__ b1, const float* __restrict__ b2,
               const float* __restrict__ ba, const float* __restrict__ bb,
               const float* __restrict__ Wc, const float* __restrict__ bc,
               float* __restrict__ dx_out, float* __restrict__ dvec_out, int Nn) {
    extern __shared__ char smem[];
    __bf16* A_lds    = (__bf16*)(smem + OFF_X);     // phase 0-2: vec tile bf16 [48][264]
    __bf16* s_lds    = (__bf16*)(smem + OFF_X);     // phase 3-4: s rows bf16 [4096][4]
    __bf16* h1_lds   = (__bf16*)(smem + OFF_X);     // phase 5-6: h1 bf16 [16][264]
    float*  v1_lds   = (float*)(smem + OFF_VEC1);   // vec1 f32 [48][260]
    float*  v2_lds   = (float*)(smem + OFF_VEC2);   // vec2 f32 [48][260]
    __bf16* t1s_all  = (__bf16*)(smem + OFF_T1S);   // phase 4: 8 waves x [16][64] bf16
    float*  vdot_lds = (float*)(smem + OFF_VDOT);   // vec_dot f32 [16][260]
    __bf16* xh_lds   = (__bf16*)(smem + OFF_XH);    // xh bf16 [16][520]

    const int tid  = threadIdx.x;
    const int wave = tid >> 5;
    const int lane = tid & 31;
    const int n0   = blockIdx.x * NPB;
    const int nvalid = min(NPB, Nn - n0);
    const float inv_sqrt2  = 0.70710678118654752f;
    const float inv_sqrt_h = 0.0625f;  // 1/sqrt(256)

    // ---- Phase 0: stage vec tile (48x256 f32 -> bf16) and x half of xh -----
    {
        const int rmax = nvalid * 3;
#pragma unroll 4
        for (int r = 0; r < 48; ++r) {
            float v = (r < rmax) ? vec[((size_t)(n0 * 3 + r)) * H + tid] : 0.f;
            A_lds[r * STA + tid] = (__bf16)v;
        }
#pragma unroll 4
        for (int i = 0; i < NPB; ++i) {
            float v = (i < nvalid) ? x[((size_t)(n0 + i)) * H + tid] : 0.f;
            xh_lds[i * STXH + tid] = (__bf16)v;
        }
    }
    __syncthreads();

    // ---- Phase A: V = vec(48x256) @ W_equi(256x512); vec1|vec2 -> f32 LDS --
    {
        const int n = lane & 15, h8 = (lane >> 4) * 8;
#pragma unroll
        for (int mt = 0; mt < 3; ++mt) {
            v8f acc[4] = {zero8(), zero8(), zero8(), zero8()};
#pragma unroll
            for (int k = 0; k < 8; ++k) {
                v16bf a = load_a_lds(A_lds + mt * 16 * STA + k * 32, STA, lane);
#pragma unroll
                for (int j = 0; j < 4; ++j) {
                    v16bf bfr = load_b_glb(WtE, H, (wave * 4 + j) * 16, k * 32, lane);
                    acc[j] = WMMA_BF16(a, bfr, acc[j]);
                }
            }
#pragma unroll
            for (int j = 0; j < 4; ++j) {
                int col = (wave * 4 + j) * 16 + n;
                float* dst = (col < H) ? v1_lds : v2_lds;
                int cc = col & (H - 1);
#pragma unroll
                for (int v = 0; v < 8; ++v)
                    dst[(mt * 16 + h8 + v) * ST48 + cc] = acc[j][v];
            }
        }
    }
    __syncthreads();

    // ---- Phase 3: vec_dot and s = (frame^T . vec1) rows ---------------------
    {
#pragma unroll 2
        for (int i = 0; i < NPB; ++i) {
            int c = tid;
            float a0 = v1_lds[(i * 3 + 0) * ST48 + c];
            float a1 = v1_lds[(i * 3 + 1) * ST48 + c];
            float a2 = v1_lds[(i * 3 + 2) * ST48 + c];
            float c0 = v2_lds[(i * 3 + 0) * ST48 + c];
            float c1 = v2_lds[(i * 3 + 1) * ST48 + c];
            float c2 = v2_lds[(i * 3 + 2) * ST48 + c];
            vdot_lds[i * ST48 + c] = (a0 * c0 + a1 * c1 + a2 * c2) * inv_sqrt_h;
            int gi = n0 + ((i < nvalid) ? i : 0);
            const float* f = frame + (size_t)gi * 9;  // uniform -> scalar loads
#pragma unroll
            for (int xd = 0; xd < 3; ++xd) {
                float sv = a0 * f[0 * 3 + xd] + a1 * f[1 * 3 + xd] + a2 * f[2 * 3 + xd];
                s_lds[(i * H + c) * 4 + xd] = (__bf16)sv;
            }
            s_lds[(i * H + c) * 4 + 3] = (__bf16)0.f;
        }
    }
    __syncthreads();

    // ---- Phase 4: frame MLP (silu(s@Wa+ba) -> silu(@Wb+bb) -> @Wc+bc) -------
    {
        __bf16* t1s = t1s_all + wave * (16 * 64);
        for (int idx = lane; idx < 256; idx += 32) {  // zero K-pad cols 48..63
            t1s[(idx >> 4) * 64 + 48 + (idx & 15)] = (__bf16)0.f;
        }
        const int n = lane & 15, h8 = (lane >> 4) * 8;
        float ba_c[3];
#pragma unroll
        for (int j = 0; j < 3; ++j) ba_c[j] = ba[j * 16 + n];
        float bb_n = (n < FH2) ? bb[n] : 0.f;
        float wc_n = (n < FH2) ? Wc[n] : 0.f;
        float bc0  = bc[0];

        for (int mt = wave; mt < 256; mt += NWAVES) {
            // layer a: A = s rows (K=3, zero-padded to 32)
            v16bf a;
#pragma unroll
            for (int j = 0; j < 16; ++j) a[j] = (__bf16)0.f;
            if (lane < 16) {
                const __bf16* p = s_lds + (mt * 16 + lane) * 4;
                a[0] = p[0]; a[1] = p[1]; a[2] = p[2]; a[3] = p[3];
            }
#pragma unroll
            for (int j = 0; j < 3; ++j) {
                v16bf bfr = load_b_glb(WtA, 32, j * 16, 0, lane);
                v8f acc = WMMA_BF16(a, bfr, zero8());
#pragma unroll
                for (int v = 0; v < 8; ++v)
                    t1s[(h8 + v) * 64 + j * 16 + n] = (__bf16)silu(acc[v] + ba_c[j]);
            }
            // layer b: K = 48 padded to 64 (two k-steps)
            v8f accb = zero8();
#pragma unroll
            for (int ks = 0; ks < 2; ++ks) {
                v16bf a2  = load_a_lds(t1s + ks * 32, 64, lane);
                v16bf bfr = load_b_glb(WtB, 64, 0, ks * 32, lane);
                accb = WMMA_BF16(a2, bfr, accb);
            }
            // layer c: dot with Wc across the 8 valid columns (xor-reduce)
#pragma unroll
            for (int v = 0; v < 8; ++v) {
                float p = silu(accb[v] + bb_n) * wc_n;
                p += __shfl_xor(p, 1);
                p += __shfl_xor(p, 2);
                p += __shfl_xor(p, 4);
                if (n == 0) {
                    int m = mt * 16 + h8 + v;
                    int i = m >> 8, c = m & (H - 1);
                    xh_lds[i * STXH + H + c] = (__bf16)(p + bc0);
                }
            }
        }
    }
    __syncthreads();

    // ---- Phase 5: h1 = silu(xh(16x512) @ W1(512x256) + b1) ------------------
    {
        const int n = lane & 15, h8 = (lane >> 4) * 8;
        float b1_c[2];
#pragma unroll
        for (int j = 0; j < 2; ++j) b1_c[j] = b1[(wave * 2 + j) * 16 + n];
        v8f acc[2] = {zero8(), zero8()};
#pragma unroll
        for (int k = 0; k < 16; ++k) {
            v16bf a = load_a_lds(xh_lds + k * 32, STXH, lane);
#pragma unroll
            for (int j = 0; j < 2; ++j) {
                v16bf bfr = load_b_glb(Wt1, 2 * H, (wave * 2 + j) * 16, k * 32, lane);
                acc[j] = WMMA_BF16(a, bfr, acc[j]);
            }
        }
#pragma unroll
        for (int j = 0; j < 2; ++j) {
            int col = (wave * 2 + j) * 16 + n;
#pragma unroll
            for (int v = 0; v < 8; ++v)
                h1_lds[(h8 + v) * STA + col] = (__bf16)silu(acc[j][v] + b1_c[j]);
        }
    }
    __syncthreads();

    // ---- Phase 6: xvh = h1(16x256) @ W2(256x768) + b2; fused epilogue -------
    {
        const int n = lane & 15, h8 = (lane >> 4) * 8;
#pragma unroll
        for (int t = 0; t < 2; ++t) {
            int ct = wave * 2 + t;
            int c  = ct * 16 + n;
            float b2_1 = b2[c], b2_2 = b2[H + c], b2_3 = b2[2 * H + c];
            v8f a1 = zero8(), a2 = zero8(), a3 = zero8();
#pragma unroll
            for (int k = 0; k < 8; ++k) {
                v16bf a = load_a_lds(h1_lds + k * 32, STA, lane);
                v16bf bf1 = load_b_glb(Wt2, H, ct * 16,          k * 32, lane);
                a1 = WMMA_BF16(a, bf1, a1);
                v16bf bf2 = load_b_glb(Wt2, H, H + ct * 16,      k * 32, lane);
                a2 = WMMA_BF16(a, bf2, a2);
                v16bf bf3 = load_b_glb(Wt2, H, 2 * H + ct * 16,  k * 32, lane);
                a3 = WMMA_BF16(a, bf3, a3);
            }
#pragma unroll
            for (int v = 0; v < 8; ++v) {
                int m = h8 + v;
                if (m < nvalid) {
                    float xv1 = a1[v] + b2_1;
                    float xv2 = a2[v] + b2_2;
                    float xv3 = a3[v] + b2_3;
                    float vd  = vdot_lds[m * ST48 + c];
                    size_t gn = (size_t)(n0 + m);
                    dx_out[gn * H + c] = (xv1 + xv2 * vd) * inv_sqrt2;
#pragma unroll
                    for (int d = 0; d < 3; ++d)
                        dvec_out[(gn * 3 + d) * H + c] =
                            xv3 * v2_lds[(m * 3 + d) * ST48 + c] * inv_sqrt_h;
                }
            }
        }
    }
}

// ---------------------------------------------------------------------------
extern "C" void kernel_launch(void* const* d_in, const int* in_sizes, int n_in,
                              void* d_out, int out_size, void* d_ws, size_t ws_size,
                              hipStream_t stream) {
    const float* x      = (const float*)d_in[0];
    const float* vec    = (const float*)d_in[1];
    const float* frame  = (const float*)d_in[2];
    const float* W_equi = (const float*)d_in[3];
    const float* W1     = (const float*)d_in[4];
    const float* b1     = (const float*)d_in[5];
    const float* W2     = (const float*)d_in[6];
    const float* b2     = (const float*)d_in[7];
    const float* Wa     = (const float*)d_in[8];
    const float* ba     = (const float*)d_in[9];
    const float* Wb     = (const float*)d_in[10];
    const float* bb     = (const float*)d_in[11];
    const float* Wc     = (const float*)d_in[12];
    const float* bc     = (const float*)d_in[13];
    const int Nn = in_sizes[0] / H;

    // Workspace: bf16 N-major transposed weights (~0.9 MB total, L2-resident).
    char* ws = (char*)d_ws;
    __bf16* WtE = (__bf16*)(ws + 0);        // [512][256]
    __bf16* Wt1 = (__bf16*)(ws + 262144);   // [256][512]
    __bf16* Wt2 = (__bf16*)(ws + 524288);   // [768][256]
    __bf16* WtA = (__bf16*)(ws + 917504);   // [48][32]  (K pad 3->32)
    __bf16* WtB = (__bf16*)(ws + 920576);   // [16][64]  (8->16 cols, K 48->64)

    auto tp = [&](const float* src, __bf16* dst, int Kin, int Nin, int Kpad, int Npad) {
        int total = Kpad * Npad;
        k_transpose_pad<<<(total + 255) / 256, 256, 0, stream>>>(src, dst, Kin, Nin, Kpad, Npad);
    };
    tp(W_equi, WtE, 256, 512, 256, 512);
    tp(W1,     Wt1, 512, 256, 512, 256);
    tp(W2,     Wt2, 256, 768, 256, 768);
    tp(Wa,     WtA, 3,   48,  32,  48);
    tp(Wb,     WtB, 48,  8,   64,  16);

    float* dx_out   = (float*)d_out;
    float* dvec_out = dx_out + (size_t)Nn * H;

    hipFuncSetAttribute((const void*)fte_fused,
                        hipFuncAttributeMaxDynamicSharedMemorySize, LDS_BYTES);
    int nblocks = (Nn + NPB - 1) / NPB;
    fte_fused<<<nblocks, THREADS, LDS_BYTES, stream>>>(
        x, vec, frame, WtE, Wt1, Wt2, WtA, WtB,
        b1, b2, ba, bb, Wc, bc, dx_out, dvec_out, Nn);
}
